// MultiHeadSelfAttention_28939489641222
// MI455X (gfx1250) — matrix operand order repaired and checked by reading, not compiled, not run on hardware
//
#include <hip/hip_runtime.h>
#include <hip/hip_bf16.h>
#include <stdint.h>

// ---------------------------------------------------------------------------
// MHA forward for MI455X (gfx1250, wave32, WMMA).
// All matmuls use v_wmma_f32_16x16x32_bf16 (bf16 inputs, fp32 accum).
// GEMMs: 128x128 block tile staged through LDS (async-to-LDS when available).
// ---------------------------------------------------------------------------

#define D_MODEL   1024
#define NUM_HEADS 16
#define HEAD_DIM  64
#define SEQ_T     4096
#define LDSS      40        // padded LDS row stride in elements (80B = 20 banks)

typedef __bf16   v16bf __attribute__((ext_vector_type(16)));
typedef float    v8f   __attribute__((ext_vector_type(8)));
typedef unsigned int u32x4 __attribute__((ext_vector_type(4)));
typedef unsigned int u32x2 __attribute__((ext_vector_type(2)));
typedef float    f32x4 __attribute__((ext_vector_type(4)));

// 16 bf16 = 32 bytes = two 128-bit chunks (trivial union: ext vectors only)
union Frag {
    v16bf v;
    u32x4 q[2];
    unsigned short u[16];
};

__device__ __forceinline__ unsigned short f32_bf16(float f) {
    unsigned int u = __float_as_uint(f);
    unsigned int r = 0x7FFFu + ((u >> 16) & 1u);   // round-to-nearest-even
    return (unsigned short)((u + r) >> 16);
}

__device__ __forceinline__ v8f v8f_zero() {
    v8f z = {0.f, 0.f, 0.f, 0.f, 0.f, 0.f, 0.f, 0.f};
    return z;
}

__device__ __forceinline__ v8f wmma_bf16(const Frag& a, const Frag& b, v8f c) {
    // (neg_a, A, neg_b, B, c_mod, C, reuse_a, reuse_b)
    return __builtin_amdgcn_wmma_f32_16x16x32_bf16(false, a.v, false, b.v,
                                                   (short)0, c, false, false);
}

// ---- CDNA5 async global->LDS path (guarded; falls back to ds_store) -------
#if __has_builtin(__builtin_amdgcn_global_load_async_to_lds_b128)
#define GEMM_ASYNC 1
// Builtin signature (from clang diagnostic): param0 = int4-vector pointer in
// the global (__device__/AS1) address space; param1 = LDS (AS3) pointer.
typedef int v4i_vs __attribute__((vector_size(16)));
typedef __attribute__((address_space(1))) v4i_vs* gptr_b128;
typedef __attribute__((address_space(3))) v4i_vs* lptr_b128;

__device__ __forceinline__ void async_ld_b128(const unsigned short* g,
                                              unsigned short* l) {
    // flat global addr == AS1 addr; ISA: LDS flat addr[31:0] == LDS offset,
    // so integer truncation to the 32-bit AS3 pointer is exact.
    __builtin_amdgcn_global_load_async_to_lds_b128(
        (gptr_b128)(uintptr_t)g, (lptr_b128)(uintptr_t)l, 0, 0);
}
#if __has_builtin(__builtin_amdgcn_s_wait_asynccnt)
#define WAIT_ASYNC(n) __builtin_amdgcn_s_wait_asynccnt(n)
#else
#define WAIT_ASYNC(n) asm volatile("s_wait_asynccnt %0" ::"i"(n) : "memory")
#endif
#else
#define GEMM_ASYNC 0
#endif

// ---------------------------------------------------------------------------
// fp32 -> bf16 conversion, 4 elements / thread (b128 load, b64 store)
// ---------------------------------------------------------------------------
__global__ void cvt_f32_to_bf16(const float* __restrict__ in,
                                unsigned short* __restrict__ out, int n4) {
    int i = blockIdx.x * blockDim.x + threadIdx.x;
    if (i >= n4) return;
    f32x4 f = ((const f32x4*)in)[i];
    u32x2 r;
    r.x = (unsigned)f32_bf16(f.x) | ((unsigned)f32_bf16(f.y) << 16);
    r.y = (unsigned)f32_bf16(f.z) | ((unsigned)f32_bf16(f.w) << 16);
    ((u32x2*)out)[i] = r;
}

// ---------------------------------------------------------------------------
// One K-step of the block tile: wave (mr,nc) computes 32x64 = 8 wmma,
// A fragments reused across the 4 N sub-tiles. All reads are ds_load_b128.
// ---------------------------------------------------------------------------
__device__ __forceinline__ void gemm_compute_step(const unsigned short* Ab,
                                                  const unsigned short* Bb,
                                                  int mr, int nc, int ml, int hi,
                                                  v8f acc[2][4]) {
    Frag af[2];
#pragma unroll
    for (int ms = 0; ms < 2; ++ms) {
        const unsigned short* ar = Ab + (mr * 32 + ms * 16 + ml) * LDSS + hi * 8;
        af[ms].q[0] = *(const u32x4*)ar;
        af[ms].q[1] = *(const u32x4*)(ar + 16);
    }
#pragma unroll
    for (int ns = 0; ns < 4; ++ns) {
        const unsigned short* br = Bb + (nc * 64 + ns * 16 + ml) * LDSS + hi * 8;
        Frag bf;
        bf.q[0] = *(const u32x4*)br;
        bf.q[1] = *(const u32x4*)(br + 16);
        acc[0][ns] = wmma_bf16(af[0], bf, acc[0][ns]);
        acc[1][ns] = wmma_bf16(af[1], bf, acc[1][ns]);
    }
}

// ---------------------------------------------------------------------------
// C(MxN) = A(MxK) * B(NxK)^T    A,B bf16 row-major (K contiguous)
// MODE 0: C bf16 row-major   MODE 1: C bf16 transposed (C[n*M+m])
// MODE 2: C fp32 row-major
// Block: 128x128, 8 waves (4 M-strips x 2 N-strips), LDS double-buffered.
// ---------------------------------------------------------------------------
template <int MODE>
__global__ __launch_bounds__(256)
void gemm_bt(const unsigned short* __restrict__ A,
             const unsigned short* __restrict__ B,
             void* __restrict__ Cv, int M, int N, int K) {
    __shared__ __align__(16) unsigned short Abuf[2][128 * LDSS];
    __shared__ __align__(16) unsigned short Bbuf[2][128 * LDSS];

    const int tid  = threadIdx.x;
    const int lane = tid & 31;
    const int wave = tid >> 5;
    const int ml = lane & 15;
    const int hi = lane >> 4;
    const int mr = wave & 3;           // M strip (4 x 32 rows)
    const int nc = wave >> 2;          // N strip (2 x 64 cols)
    const int m0 = blockIdx.x * 128;
    const int n0 = blockIdx.y * 128;

    // Staging map: 512 16B-chunks per tile; thread handles chunk tid and tid+256.
    const int srow  = tid >> 2;              // 0..63 (and +64)
    const int skoff = (tid & 3) * 8;         // 0,8,16,24 elements
    const unsigned short* Ag  = A + (size_t)(m0 + srow) * K + skoff;
    const unsigned short* Ag2 = A + (size_t)(m0 + srow + 64) * K + skoff;
    const unsigned short* Bg  = B + (size_t)(n0 + srow) * K + skoff;
    const unsigned short* Bg2 = B + (size_t)(n0 + srow + 64) * K + skoff;
    const int loff  = srow * LDSS + skoff;
    const int loff2 = (srow + 64) * LDSS + skoff;

    v8f acc[2][4];
#pragma unroll
    for (int ms = 0; ms < 2; ++ms)
#pragma unroll
        for (int ns = 0; ns < 4; ++ns) acc[ms][ns] = v8f_zero();

    const int nt = K / 32;

#if GEMM_ASYNC
    // ---- async double-buffered pipeline (ASYNCcnt) ----
    {
        async_ld_b128(Ag, &Abuf[0][loff]);
        async_ld_b128(Ag2, &Abuf[0][loff2]);
        async_ld_b128(Bg, &Bbuf[0][loff]);
        async_ld_b128(Bg2, &Bbuf[0][loff2]);
    }
    for (int t = 0; t < nt; ++t) {
        if (t + 1 < nt) {
            const int kn = (t + 1) * 32;
            unsigned short* An = Abuf[(t + 1) & 1];
            unsigned short* Bn = Bbuf[(t + 1) & 1];
            async_ld_b128(Ag + kn, An + loff);
            async_ld_b128(Ag2 + kn, An + loff2);
            async_ld_b128(Bg + kn, Bn + loff);
            async_ld_b128(Bg2 + kn, Bn + loff2);
            WAIT_ASYNC(4);           // in-order: current tile's 4 ops complete
        } else {
            WAIT_ASYNC(0);
        }
        __syncthreads();             // staged tile visible to all waves
        gemm_compute_step(Abuf[t & 1], Bbuf[t & 1], mr, nc, ml, hi, acc);
        __syncthreads();             // all reads done before buffer reuse
    }
#else
    // ---- fallback: register prefetch + ds_store_b128 double buffer ----
    u32x4 ra = *(const u32x4*)Ag;
    u32x4 ra2 = *(const u32x4*)Ag2;
    u32x4 rb = *(const u32x4*)Bg;
    u32x4 rb2 = *(const u32x4*)Bg2;
    for (int t = 0; t < nt; ++t) {
        unsigned short* Ab = Abuf[t & 1];
        unsigned short* Bb = Bbuf[t & 1];
        *(u32x4*)(Ab + loff) = ra;
        *(u32x4*)(Ab + loff2) = ra2;
        *(u32x4*)(Bb + loff) = rb;
        *(u32x4*)(Bb + loff2) = rb2;
        if (t + 1 < nt) {
            const int kn = (t + 1) * 32;
            ra = *(const u32x4*)(Ag + kn);
            ra2 = *(const u32x4*)(Ag2 + kn);
            rb = *(const u32x4*)(Bg + kn);
            rb2 = *(const u32x4*)(Bg2 + kn);
        }
        __syncthreads();
        gemm_compute_step(Ab, Bb, mr, nc, ml, hi, acc);
        // no trailing barrier needed: double-buffered, next stores hit buf[t^1]
    }
#endif

    // Epilogue: acc[ms][ns][i] -> (row = m0+mr*32+ms*16+i+8*hi,
    //                              col = n0+nc*64+ns*16+ml)
#pragma unroll
    for (int ms = 0; ms < 2; ++ms) {
#pragma unroll
        for (int ns = 0; ns < 4; ++ns) {
#pragma unroll
            for (int i = 0; i < 8; ++i) {
                const int row = m0 + mr * 32 + ms * 16 + i + 8 * hi;
                const int col = n0 + nc * 64 + ns * 16 + ml;
                if (MODE == 2) {
                    ((float*)Cv)[(size_t)row * N + col] = acc[ms][ns][i];
                } else if (MODE == 1) {
                    ((unsigned short*)Cv)[(size_t)col * M + row] =
                        f32_bf16(acc[ms][ns][i]);
                } else {
                    ((unsigned short*)Cv)[(size_t)row * N + col] =
                        f32_bf16(acc[ms][ns][i]);
                }
            }
        }
    }
}

// ---------------------------------------------------------------------------
// Flash attention, causal. One wave = 16 query rows of one head.
// Q,K: (T, D_MODEL) bf16 row-major.  Vt: (D_MODEL, T) bf16 (time contiguous).
// O:   (T, D_MODEL) bf16 row-major.
// Key tile = 32 keys: S = 2 C-tiles (4 wmma), P staged via LDS -> one
// A-fragment, O update = 4 wmma (one per 16-wide d slice).
// ---------------------------------------------------------------------------
__global__ __launch_bounds__(256)
void attn_fwd(const unsigned short* __restrict__ Q,
              const unsigned short* __restrict__ Km,
              const unsigned short* __restrict__ Vt,
              unsigned short* __restrict__ O) {
    __shared__ __align__(16) unsigned short pstage[8 * 16 * 32];  // 1KB/wave

    const int lane = threadIdx.x & 31;
    const int wave = threadIdx.x >> 5;
    const int q0 = blockIdx.x * 128 + wave * 16;
    const int h  = blockIdx.y;
    const int ml = lane & 15;
    const int hi = lane >> 4;
    const float scale = 0.125f;                    // 1/sqrt(64)
    const float NEG = -3.0e38f;                    // finite -inf sentinel

    // Q A-fragments (d=64 -> two K=32 fragments), reused for every key tile
    Frag aq[2];
#pragma unroll
    for (int kt = 0; kt < 2; ++kt) {
        const unsigned short* qrow =
            Q + (size_t)(q0 + ml) * D_MODEL + h * HEAD_DIM + kt * 32 + hi * 8;
        aq[kt].q[0] = *(const u32x4*)qrow;
        aq[kt].q[1] = *(const u32x4*)(qrow + 16);
    }

    float rmax[8], rsum[8];
    v8f o[4];
#pragma unroll
    for (int i = 0; i < 8; ++i) { rmax[i] = NEG; rsum[i] = 0.f; }
#pragma unroll
    for (int t = 0; t < 4; ++t) o[t] = v8f_zero();

    unsigned short* ps = pstage + wave * (16 * 32);

    // keys needed: 0 .. q0+15  ->  ceil((q0+16)/32) tiles of 32
    const int ntiles = (q0 + 47) >> 5;
    for (int ti = 0; ti < ntiles; ++ti) {
        const int sbase = ti * 32;

        if (ti + 1 < ntiles) {   // speculative next-tile prefetch (L2 warm)
            __builtin_prefetch(Km + (size_t)(sbase + 32 + ml) * D_MODEL +
                                   h * HEAD_DIM, 0, 1);
            __builtin_prefetch(Vt + (size_t)(h * HEAD_DIM + ml) * SEQ_T +
                                   sbase + 32, 0, 1);
        }

        // ---- S = Q * K^T  (two 16x16 column tiles) ----
        v8f s2[2];
        s2[0] = v8f_zero(); s2[1] = v8f_zero();
#pragma unroll
        for (int nt = 0; nt < 2; ++nt) {
#pragma unroll
            for (int kt = 0; kt < 2; ++kt) {
                const unsigned short* krow =
                    Km + (size_t)(sbase + nt * 16 + ml) * D_MODEL +
                    h * HEAD_DIM + kt * 32 + hi * 8;
                Frag b;
                b.q[0] = *(const u32x4*)krow;
                b.q[1] = *(const u32x4*)(krow + 16);
                s2[nt] = wmma_bf16(aq[kt], b, s2[nt]);
            }
        }

        // ---- online softmax (row r lives in 16 lanes of one half-wave) ----
#pragma unroll
        for (int i = 0; i < 8; ++i) {
            const int qr = q0 + i + 8 * hi;
            float v0 = s2[0][i] * scale;
            float v1 = s2[1][i] * scale;
            if (sbase + ml > qr)      v0 = NEG;    // causal mask
            if (sbase + 16 + ml > qr) v1 = NEG;

            float m = fmaxf(v0, v1);
#pragma unroll
            for (int off = 8; off >= 1; off >>= 1)
                m = fmaxf(m, __shfl_xor(m, off, 16));

            const float nm = fmaxf(rmax[i], m);
            const float corr = __expf(rmax[i] - nm);
            rmax[i] = nm;

            const float p0 = __expf(v0 - nm);
            const float p1 = __expf(v1 - nm);
            float rs = p0 + p1;
#pragma unroll
            for (int off = 8; off >= 1; off >>= 1)
                rs += __shfl_xor(rs, off, 16);
            rsum[i] = rsum[i] * corr + rs;
#pragma unroll
            for (int dt = 0; dt < 4; ++dt) o[dt][i] *= corr;

            // stage P (C-layout -> row-major 16x32 bf16 in LDS)
            const int r = i + 8 * hi;
            ps[r * 32 + ml]      = f32_bf16(p0);
            ps[r * 32 + 16 + ml] = f32_bf16(p1);
        }

        // ---- read P back as A-fragment (DS ops are in-order within a wave) --
        Frag ap;
        ap.q[0] = *(const u32x4*)(ps + ml * 32 + hi * 8);
        ap.q[1] = *(const u32x4*)(ps + ml * 32 + hi * 8 + 16);

        // ---- O += P * V  (V transposed: fixed d-col, contiguous in time) ----
#pragma unroll
        for (int dt = 0; dt < 4; ++dt) {
            const unsigned short* vrow =
                Vt + (size_t)(h * HEAD_DIM + dt * 16 + ml) * SEQ_T + sbase + hi * 8;
            Frag bv;
            bv.q[0] = *(const u32x4*)vrow;
            bv.q[1] = *(const u32x4*)(vrow + 16);
            o[dt] = wmma_bf16(ap, bv, o[dt]);
        }
    }

    // ---- normalize & store bf16 (T, D_MODEL) ----
#pragma unroll
    for (int dt = 0; dt < 4; ++dt) {
#pragma unroll
        for (int i = 0; i < 8; ++i) {
            const int qr = q0 + i + 8 * hi;
            O[(size_t)qr * D_MODEL + h * HEAD_DIM + dt * 16 + ml] =
                f32_bf16(o[dt][i] / rsum[i]);
        }
    }
}

// ---------------------------------------------------------------------------
// Launch: cvt x + 4 weights -> bf16; Q,K,V projections; flash attention;
// output projection (fp32 out). Workspace budget: 40 MB.
// ---------------------------------------------------------------------------
extern "C" void kernel_launch(void* const* d_in, const int* in_sizes, int n_in,
                              void* d_out, int out_size, void* d_ws, size_t ws_size,
                              hipStream_t stream) {
    const float* x  = (const float*)d_in[0];
    const float* Wq = (const float*)d_in[1];
    const float* Wk = (const float*)d_in[2];
    const float* Wv = (const float*)d_in[3];
    const float* Wo = (const float*)d_in[4];

    uint8_t* ws = (uint8_t*)d_ws;
    const size_t MB = 1024 * 1024;
    // xb region (8MB) is reused for the attention output Ob after QKV GEMMs.
    unsigned short* xb  = (unsigned short*)(ws + 0);
    unsigned short* Ob  = (unsigned short*)(ws + 0);
    unsigned short* Wqb = (unsigned short*)(ws + 8 * MB);
    unsigned short* Wkb = (unsigned short*)(ws + 10 * MB);
    unsigned short* Wvb = (unsigned short*)(ws + 12 * MB);
    unsigned short* Wob = (unsigned short*)(ws + 14 * MB);
    unsigned short* Qb  = (unsigned short*)(ws + 16 * MB);
    unsigned short* Kb  = (unsigned short*)(ws + 24 * MB);
    unsigned short* Vt  = (unsigned short*)(ws + 32 * MB);  // (D_MODEL, T)

    const int nx4 = SEQ_T * D_MODEL / 4;
    const int nw4 = D_MODEL * D_MODEL / 4;
    cvt_f32_to_bf16<<<nx4 / 256, 256, 0, stream>>>(x,  xb,  nx4);
    cvt_f32_to_bf16<<<nw4 / 256, 256, 0, stream>>>(Wq, Wqb, nw4);
    cvt_f32_to_bf16<<<nw4 / 256, 256, 0, stream>>>(Wk, Wkb, nw4);
    cvt_f32_to_bf16<<<nw4 / 256, 256, 0, stream>>>(Wv, Wvb, nw4);
    cvt_f32_to_bf16<<<nw4 / 256, 256, 0, stream>>>(Wo, Wob, nw4);

    dim3 g(SEQ_T / 128, D_MODEL / 128);      // (32, 8)
    gemm_bt<0><<<g, 256, 0, stream>>>(xb, Wqb, Qb, SEQ_T, D_MODEL, D_MODEL);
    gemm_bt<0><<<g, 256, 0, stream>>>(xb, Wkb, Kb, SEQ_T, D_MODEL, D_MODEL);
    gemm_bt<1><<<g, 256, 0, stream>>>(xb, Wvb, Vt, SEQ_T, D_MODEL, D_MODEL);

    attn_fwd<<<dim3(SEQ_T / 128, NUM_HEADS), 256, 0, stream>>>(Qb, Kb, Vt, Ob);

    gemm_bt<2><<<g, 256, 0, stream>>>(Ob, Wob, (float*)d_out, SEQ_T, D_MODEL, D_MODEL);
}
